// MultiHeadSelfAttention_21706764714421
// MI455X (gfx1250) — matrix operand order repaired and checked by reading, not compiled, not run on hardware
//
#include <hip/hip_runtime.h>
#include <hip/hip_bf16.h>
#include <math.h>

// ---------------------------------------------------------------------------
// MultiHeadSelfAttention for MI455X (gfx1250, wave32, WMMA + async-LDS)
// B=2, T=2048, C=1024, H=16, hd=64
// ---------------------------------------------------------------------------

typedef __attribute__((ext_vector_type(16))) __bf16 v16bf;
typedef __attribute__((ext_vector_type(8)))  float  v8f;

#define WMMA_BF16(a, b, c) \
  __builtin_amdgcn_wmma_f32_16x16x32_bf16(false, (a), false, (b), (short)0, (c), false, false)

// ---- async global->LDS (CDNA5 GLOBAL_LOAD_ASYNC_TO_LDS, ASYNCcnt) ---------

// Generic (flat) shared-memory addresses map to LDS via addr[31:0] (ISA 10.2),
// so truncating the generic pointer yields the LDS byte offset the async op wants.
static __device__ __forceinline__ uint32_t lds_off(const void* p) {
  return (uint32_t)(uintptr_t)p;
}

// Per-lane: LDS[ldsoff] = MEM[gaddr] (16 bytes), tracked by ASYNCcnt.
static __device__ __forceinline__ void async_load_b128(uint32_t ldsoff, const void* gaddr) {
  asm volatile("global_load_async_to_lds_b128 %0, %1, off"
               :
               : "v"(ldsoff), "v"((uint64_t)(uintptr_t)gaddr)
               : "memory");
}

static __device__ __forceinline__ void wait_async0() {
  asm volatile("s_wait_asynccnt 0" ::: "memory");
}

// ---- fragment loaders (contiguous k = 16*(lane/16) + s, wave32) --------------

// A-matrix 16x32 bf16, base at [0][0], row stride ld (elems).
static __device__ __forceinline__ v16bf load_a_frag(const __bf16* base, int ld, int lane) {
  const int m  = lane & 15;
  const int kb = (lane >> 4) << 4;
  v16bf a;
#pragma unroll
  for (int i = 0; i < 8; ++i) {
    const int k = kb + 2 * i;
    a[2 * i]     = base[m * ld + k];
    a[2 * i + 1] = base[m * ld + k + 1];
  }
  return a;
}

// B 32x16 where memory holds B^T: base is a [n][k] tile, row stride ld.
static __device__ __forceinline__ v16bf load_bt_frag(const __bf16* base, int ld, int lane) {
  const int n  = lane & 15;
  const int kb = (lane >> 4) << 4;
  v16bf b;
#pragma unroll
  for (int i = 0; i < 8; ++i) {
    b[2 * i]     = base[n * ld + kb + 2 * i];
    b[2 * i + 1] = base[n * ld + kb + 2 * i + 1];
  }
  return b;
}

// B 32x16 where memory holds B directly: base is a [k][n] tile, row stride ld.
static __device__ __forceinline__ v16bf load_bn_frag(const __bf16* base, int ld, int lane) {
  const int n  = lane & 15;
  const int kb = (lane >> 4) << 4;
  v16bf b;
#pragma unroll
  for (int i = 0; i < 8; ++i) {
    b[2 * i]     = base[(kb + 2 * i) * ld + n];
    b[2 * i + 1] = base[(kb + 2 * i + 1) * ld + n];
  }
  return b;
}

static __device__ __forceinline__ v8f zero8() {
  v8f z;
#pragma unroll
  for (int e = 0; e < 8; ++e) z[e] = 0.0f;
  return z;
}

// ---- fp32 -> bf16 cast -----------------------------------------------------
__global__ void mhsa_cvt_bf16(const float* __restrict__ src, __bf16* __restrict__ dst, int n) {
  int i = blockIdx.x * blockDim.x + threadIdx.x;
  const int stride = gridDim.x * blockDim.x;
  for (; i < n; i += stride) dst[i] = (__bf16)src[i];
}

// ---- QKV GEMM: qkv = x @ W_qkv^T + b ; scatter to q/k/v [B][H][T][hd] ------
// M=4096, N=3072, K=1024.  WG tile 64x256, 8 waves, each wave 32x64 (2x4 frags)
// Double-buffered LDS fed by async-to-LDS loads.
__global__ __launch_bounds__(256) void mhsa_gemm_qkv(
    const __bf16* __restrict__ A,   // [M][K] = x
    const __bf16* __restrict__ W,   // [N][K] = W_qkv
    const float*  __restrict__ bias,
    __bf16* __restrict__ qh, __bf16* __restrict__ kh, __bf16* __restrict__ vh) {
  constexpr int K = 1024;
  __shared__ __align__(16) __bf16 As[2][64 * 32];
  __shared__ __align__(16) __bf16 Bs[2][256 * 32];
  const int tid  = threadIdx.x;
  const int lane = tid & 31;
  const int wave = tid >> 5;
  const int wm   = wave >> 2;   // 0..1
  const int wn   = wave & 3;    // 0..3
  const int mBase = blockIdx.y * 64;
  const int nBase = blockIdx.x * 256;

  v8f acc[2][4];
#pragma unroll
  for (int i = 0; i < 2; ++i)
#pragma unroll
    for (int j = 0; j < 4; ++j) acc[i][j] = zero8();

  const int lrow = tid >> 2;
  const int lcol = (tid & 3) * 8;

  auto issue_tile = [&](int k0, int buf) {
    async_load_b128(lds_off(&As[buf][lrow * 32 + lcol]),
                    &A[(size_t)(mBase + lrow) * K + k0 + lcol]);
#pragma unroll
    for (int j = 0; j < 4; ++j) {
      const int n = lrow + j * 64;
      async_load_b128(lds_off(&Bs[buf][n * 32 + lcol]),
                      &W[(size_t)(nBase + n) * K + k0 + lcol]);
    }
  };

  issue_tile(0, 0);
  wait_async0();
  __syncthreads();

  for (int it = 0; it < K / 32; ++it) {
    if (it + 1 < K / 32) issue_tile((it + 1) * 32, (it + 1) & 1);

    const __bf16* as = As[it & 1];
    const __bf16* bs = Bs[it & 1];
    v16bf afr[2], bfr[4];
#pragma unroll
    for (int mi = 0; mi < 2; ++mi)
      afr[mi] = load_a_frag(&as[(wm * 32 + mi * 16) * 32], 32, lane);
#pragma unroll
    for (int nj = 0; nj < 4; ++nj)
      bfr[nj] = load_bt_frag(&bs[(wn * 64 + nj * 16) * 32], 32, lane);
#pragma unroll
    for (int mi = 0; mi < 2; ++mi)
#pragma unroll
      for (int nj = 0; nj < 4; ++nj)
        acc[mi][nj] = WMMA_BF16(afr[mi], bfr[nj], acc[mi][nj]);

    wait_async0();
    __syncthreads();
  }

  // epilogue: n -> (head, q/k/v, d); row m -> (b, t)
  const int half = lane >> 4;
  const int nl   = lane & 15;
#pragma unroll
  for (int mi = 0; mi < 2; ++mi) {
#pragma unroll
    for (int nj = 0; nj < 4; ++nj) {
      const int n  = nBase + wn * 64 + nj * 16 + nl;
      const float bv = bias[n];
      const int h = n / 192;
      const int rr = n % 192;
      const int which = rr >> 6;
      const int d = rr & 63;
      __bf16* dst = (which == 0) ? qh : (which == 1) ? kh : vh;
      const float scale = (which == 0) ? 0.125f : 1.0f;  // 1/sqrt(hd)
#pragma unroll
      for (int r = 0; r < 8; ++r) {
        const int m  = mBase + wm * 32 + mi * 16 + r + 8 * half;
        const int b_ = m >> 11;       // / 2048
        const int t  = m & 2047;
        const float val = (acc[mi][nj][r] + bv) * scale;
        dst[(((size_t)(b_ * 16 + h)) * 2048 + t) * 64 + d] = (__bf16)val;
      }
    }
  }
}

// ---- streaming (flash) attention ------------------------------------------
// grid = B*H*(T/64) = 1024 WGs, 128 threads (4 waves). Each wave owns 16 q rows.
// K/V tiles double-buffered via async-to-LDS.
__global__ __launch_bounds__(128) void mhsa_attn(
    const __bf16* __restrict__ q, const __bf16* __restrict__ k,
    const __bf16* __restrict__ v, __bf16* __restrict__ o) {
  __shared__ __align__(16) __bf16 Qs[64 * 64];
  __shared__ __align__(16) __bf16 Ks[2][64 * 64];
  __shared__ __align__(16) __bf16 Vs[2][64 * 64];
  __shared__ __align__(16) __bf16 Ps[4][16 * 64];

  const int tid  = threadIdx.x;
  const int lane = tid & 31;
  const int wave = tid >> 5;
  const int half = lane >> 4;
  const int nl   = lane & 15;
  const int qb   = blockIdx.x & 31;   // T/64
  const int bh   = blockIdx.x >> 5;   // b*16 + h
  const size_t headoff = (size_t)bh * 2048 * 64;

  auto issue_kv = [&](int kb, int buf) {
    const __bf16* kp = k + headoff + (size_t)kb * 64 * 64;
    const __bf16* vp = v + headoff + (size_t)kb * 64 * 64;
#pragma unroll
    for (int i = 0; i < 4; ++i) {
      async_load_b128(lds_off(&Ks[buf][(tid + i * 128) * 8]), &kp[(tid + i * 128) * 8]);
      async_load_b128(lds_off(&Vs[buf][(tid + i * 128) * 8]), &vp[(tid + i * 128) * 8]);
    }
  };

  // Q tile (contiguous 64x64) + first K/V block
  const __bf16* qp = q + headoff + (size_t)qb * 64 * 64;
#pragma unroll
  for (int i = 0; i < 4; ++i)
    async_load_b128(lds_off(&Qs[(tid + i * 128) * 8]), &qp[(tid + i * 128) * 8]);
  issue_kv(0, 0);
  wait_async0();
  __syncthreads();

  v16bf aq[2];
  aq[0] = load_a_frag(&Qs[wave * 16 * 64], 64, lane);
  aq[1] = load_a_frag(&Qs[wave * 16 * 64 + 32], 64, lane);

  float mrow[8], lsum[8];
  v8f O[4];
#pragma unroll
  for (int r = 0; r < 8; ++r) { mrow[r] = -1e30f; lsum[r] = 0.0f; }
#pragma unroll
  for (int j = 0; j < 4; ++j) O[j] = zero8();

  for (int kb = 0; kb < 32; ++kb) {
    if (kb + 1 < 32) issue_kv(kb + 1, (kb + 1) & 1);

    const __bf16* ks = Ks[kb & 1];
    const __bf16* vs = Vs[kb & 1];

    // S = Q K^T  (16 rows x 64 keys per wave)
    v8f S[4];
#pragma unroll
    for (int j = 0; j < 4; ++j) {
      S[j] = zero8();
#pragma unroll
      for (int kss = 0; kss < 2; ++kss) {
        v16bf bk = load_bt_frag(&ks[(j * 16) * 64 + kss * 32], 64, lane);
        S[j] = WMMA_BF16(aq[kss], bk, S[j]);
      }
    }

    // online softmax, rows live across 16 lanes of each half-wave
#pragma unroll
    for (int r = 0; r < 8; ++r) {
      float mx = S[0][r];
#pragma unroll
      for (int j = 1; j < 4; ++j) mx = fmaxf(mx, S[j][r]);
#pragma unroll
      for (int msk = 1; msk < 16; msk <<= 1) mx = fmaxf(mx, __shfl_xor(mx, msk, 32));
      const float mnew  = fmaxf(mrow[r], mx);
      const float scale = __expf(mrow[r] - mnew);
      float rs = 0.0f;
#pragma unroll
      for (int j = 0; j < 4; ++j) {
        const float p = __expf(S[j][r] - mnew);
        S[j][r] = p;
        rs += p;
      }
#pragma unroll
      for (int msk = 1; msk < 16; msk <<= 1) rs += __shfl_xor(rs, msk, 32);
      lsum[r] = lsum[r] * scale + rs;
      mrow[r] = mnew;
#pragma unroll
      for (int j = 0; j < 4; ++j) {
        O[j][r] *= scale;
        Ps[wave][(r + 8 * half) * 64 + j * 16 + nl] = (__bf16)S[j][r];
      }
    }
    __syncthreads();   // Ps visibility across the wave's LDS round-trip

    // O += P @ V
#pragma unroll
    for (int kss = 0; kss < 2; ++kss) {
      v16bf ap = load_a_frag(&Ps[wave][kss * 32], 64, lane);
#pragma unroll
      for (int j = 0; j < 4; ++j) {
        v16bf bvf = load_bn_frag(&vs[(kss * 32) * 64 + j * 16], 64, lane);
        O[j] = WMMA_BF16(ap, bvf, O[j]);
      }
    }

    wait_async0();
    __syncthreads();
  }

  // normalize and write attn output as [B][T][C] bf16
  const int h  = bh & 15;
  const int b_ = bh >> 4;
#pragma unroll
  for (int j = 0; j < 4; ++j) {
#pragma unroll
    for (int r = 0; r < 8; ++r) {
      const int t = qb * 64 + wave * 16 + r + 8 * half;
      const int c = h * 64 + j * 16 + nl;
      o[((size_t)(b_ * 2048) + t) * 1024 + c] = (__bf16)(O[j][r] / lsum[r]);
    }
  }
}

// ---- output projection: out = attn @ W_proj^T + b_proj  (fp32 out) ---------
__global__ __launch_bounds__(256) void mhsa_gemm_proj(
    const __bf16* __restrict__ A,   // [4096][1024]
    const __bf16* __restrict__ W,   // [1024][1024]
    const float*  __restrict__ bias,
    float* __restrict__ out) {
  constexpr int K = 1024;
  constexpr int N = 1024;
  __shared__ __align__(16) __bf16 As[2][64 * 32];
  __shared__ __align__(16) __bf16 Bs[2][256 * 32];
  const int tid  = threadIdx.x;
  const int lane = tid & 31;
  const int wave = tid >> 5;
  const int wm   = wave >> 2;
  const int wn   = wave & 3;
  const int mBase = blockIdx.y * 64;
  const int nBase = blockIdx.x * 256;

  v8f acc[2][4];
#pragma unroll
  for (int i = 0; i < 2; ++i)
#pragma unroll
    for (int j = 0; j < 4; ++j) acc[i][j] = zero8();

  const int lrow = tid >> 2;
  const int lcol = (tid & 3) * 8;

  auto issue_tile = [&](int k0, int buf) {
    async_load_b128(lds_off(&As[buf][lrow * 32 + lcol]),
                    &A[(size_t)(mBase + lrow) * K + k0 + lcol]);
#pragma unroll
    for (int j = 0; j < 4; ++j) {
      const int n = lrow + j * 64;
      async_load_b128(lds_off(&Bs[buf][n * 32 + lcol]),
                      &W[(size_t)(nBase + n) * K + k0 + lcol]);
    }
  };

  issue_tile(0, 0);
  wait_async0();
  __syncthreads();

  for (int it = 0; it < K / 32; ++it) {
    if (it + 1 < K / 32) issue_tile((it + 1) * 32, (it + 1) & 1);

    const __bf16* as = As[it & 1];
    const __bf16* bs = Bs[it & 1];
    v16bf afr[2], bfr[4];
#pragma unroll
    for (int mi = 0; mi < 2; ++mi)
      afr[mi] = load_a_frag(&as[(wm * 32 + mi * 16) * 32], 32, lane);
#pragma unroll
    for (int nj = 0; nj < 4; ++nj)
      bfr[nj] = load_bt_frag(&bs[(wn * 64 + nj * 16) * 32], 32, lane);
#pragma unroll
    for (int mi = 0; mi < 2; ++mi)
#pragma unroll
      for (int nj = 0; nj < 4; ++nj)
        acc[mi][nj] = WMMA_BF16(afr[mi], bfr[nj], acc[mi][nj]);

    wait_async0();
    __syncthreads();
  }

  const int half = lane >> 4;
  const int nl   = lane & 15;
#pragma unroll
  for (int mi = 0; mi < 2; ++mi) {
#pragma unroll
    for (int nj = 0; nj < 4; ++nj) {
      const int n = nBase + wn * 64 + nj * 16 + nl;
      const float bv = bias[n];
#pragma unroll
      for (int r = 0; r < 8; ++r) {
        const int m = mBase + wm * 32 + mi * 16 + r + 8 * half;
        out[(size_t)m * N + n] = acc[mi][nj][r] + bv;
      }
    }
  }
}

// ---------------------------------------------------------------------------
extern "C" void kernel_launch(void* const* d_in, const int* in_sizes, int n_in,
                              void* d_out, int out_size, void* d_ws, size_t ws_size,
                              hipStream_t stream) {
  const float* x      = (const float*)d_in[0];
  const float* W_qkv  = (const float*)d_in[1];
  const float* b_qkv  = (const float*)d_in[2];
  const float* W_proj = (const float*)d_in[3];
  const float* b_proj = (const float*)d_in[4];
  float* out = (float*)d_out;

  constexpr size_t NX    = 2ull * 2048 * 1024;  // 4,194,304
  constexpr size_t NWQKV = 3ull * 1024 * 1024;  // 3,145,728
  constexpr size_t NWPRJ = 1ull * 1024 * 1024;  // 1,048,576

  char* ws = (char*)d_ws;
  size_t off = 0;
  auto alloc = [&](size_t bytes) -> void* {
    void* p = ws + off;
    off += (bytes + 255) & ~(size_t)255;
    return p;
  };
  __bf16* xb     = (__bf16*)alloc(NX * 2);
  __bf16* wqkvb  = (__bf16*)alloc(NWQKV * 2);
  __bf16* wprojb = (__bf16*)alloc(NWPRJ * 2);
  __bf16* qh     = (__bf16*)alloc(NX * 2);
  __bf16* kh     = (__bf16*)alloc(NX * 2);
  __bf16* vh     = (__bf16*)alloc(NX * 2);
  __bf16* ao     = (__bf16*)alloc(NX * 2);

  mhsa_cvt_bf16<<<2048, 256, 0, stream>>>(x, xb, (int)NX);
  mhsa_cvt_bf16<<<2048, 256, 0, stream>>>(W_qkv, wqkvb, (int)NWQKV);
  mhsa_cvt_bf16<<<2048, 256, 0, stream>>>(W_proj, wprojb, (int)NWPRJ);

  mhsa_gemm_qkv<<<dim3(12, 64), 256, 0, stream>>>(xb, wqkvb, b_qkv, qh, kh, vh);

  mhsa_attn<<<1024, 128, 0, stream>>>(qh, kh, vh, ao);

  mhsa_gemm_proj<<<dim3(4, 64), 256, 0, stream>>>(ao, wprojb, b_proj, out);
}
